// MultiHeadAttention_8057358648309
// MI455X (gfx1250) — matrix operand order repaired and checked by reading, not compiled, not run on hardware
//
#include <hip/hip_runtime.h>
#include <hip/hip_bf16.h>

// ---- MI455X / gfx1250 fused multi-head attention, pipelined TDM + WMMA ----
// Pass 0: one-shot f32->f16 conversion of x / Wqkv / Wp into workspace.
// Kernel A: fused QKV projection + causal attention per (batch, head).
//           Whole x slab + double-buffered weight tiles staged by the Tensor
//           Data Mover; s_wait_tensorcnt N overlaps DMA with WMMA.
// Kernel B: output projection GEMM + bias, double-buffered TDM staging.

typedef __attribute__((ext_vector_type(16))) _Float16 v16h;
typedef __attribute__((ext_vector_type(8)))  _Float16 v8h;
typedef __attribute__((ext_vector_type(4)))  _Float16 v4h;
typedef __attribute__((ext_vector_type(8)))  float    v8f;
typedef __attribute__((ext_vector_type(4)))  unsigned v4u;
typedef __attribute__((ext_vector_type(8)))  unsigned v8u;

#define N_EMBD 384
#define TSEQ   64
#define HN     6
#define HS     64
#define NCHUNK 12   // 384 / 32

__device__ __forceinline__ v8f wmma_f16(v16h a, v16h b, v8f c) {
    return __builtin_amdgcn_wmma_f32_16x16x32_f16(
        false, a, false, b, (short)0, c, false, false);
}

// ---- Tensor Data Mover: 2D tile (f16) global -> LDS ------------------------
// D# group0: [1:0]=count=1, [63:32]=lds_addr, [120:64]=global_addr, [127:126]=type=2
// D# group1: data_size=1 (2B), tensor_dim0/1, tile_dim0/1, tensor_dim0_stride.
__device__ __forceinline__ void tdm_load_tile_f16(
    const _Float16* gbase, unsigned lds_byte_off,
    unsigned tile_x, unsigned tile_y,
    unsigned tdim0, unsigned tdim1, unsigned stride_elems)
{
    unsigned long long ga = (unsigned long long)(uintptr_t)gbase;
    v4u g0;
    g0[0] = 0x1u;                                   // count=1 (user descriptor)
    g0[1] = lds_byte_off;                           // lds_addr
    g0[2] = (unsigned)ga;                           // global_addr[31:0]
    g0[3] = ((unsigned)(ga >> 32) & 0x01FFFFFFu)    // global_addr[56:32]
            | (2u << 30);                           // type=2 ("image")
    v8u g1;
    g1[0] = 0x00010000u;                            // data_size=1 -> 2 bytes
    g1[1] = (tdim0 & 0xFFFFu) << 16;                // tensor_dim0 @ bits79:48
    g1[2] = (tdim0 >> 16) | ((tdim1 & 0xFFFFu) << 16);
    g1[3] = (tdim1 >> 16) | (tile_x << 16);         // tile_dim0 @ bits127:112
    g1[4] = tile_y & 0xFFFFu;                       // tile_dim1 (tile_dim2=0)
    g1[5] = stride_elems;                           // tensor_dim0_stride[31:0]
    g1[6] = 0u;
    g1[7] = 0u;
    asm volatile("tensor_load_to_lds %0, %1" :: "s"(g0), "s"(g1) : "memory");
}

__device__ __forceinline__ unsigned lds_addr_of(const void* p) {
    // Flat LDS addresses are {SHARED_BASE[63:32], lds_offset[31:0]}.
    return (unsigned)(unsigned long long)(uintptr_t)p;
}

// A fragment (16x32 f16), row-major LDS, ld = row stride in halves.
__device__ __forceinline__ v16h load_frag_a(const _Float16* lds, int row_base,
                                            int k_base, int ld) {
    int lane = threadIdx.x & 31;
    int r  = row_base + (lane & 15);
    int k0 = k_base + ((lane >> 4) << 3);
    const _Float16* p = lds + r * ld + k0;
    union { v16h v; v8h h[2]; } u;
    u.h[0] = *(const v8h*)(p);
    u.h[1] = *(const v8h*)(p + 16);
    return u.v;
}

// B fragment (32x16 f16, KxN), stored n-major in LDS (row = n, k contiguous).
__device__ __forceinline__ v16h load_frag_b(const _Float16* lds, int n_base,
                                            int k_base, int ld) {
    int lane = threadIdx.x & 31;
    int n  = n_base + (lane & 15);
    int k0 = k_base + ((lane >> 4) << 3);
    const _Float16* p = lds + n * ld + k0;
    union { v16h v; v8h h[2]; } u;
    u.h[0] = *(const v8h*)(p);
    u.h[1] = *(const v8h*)(p + 16);
    return u.v;
}

// ---------------- Pass 0: f32 -> f16 convert (4 elems/thread) ---------------
__global__ __launch_bounds__(256) void cvt_f16_kernel(
    const float* __restrict__ in, _Float16* __restrict__ out, int n4)
{
    int i = blockIdx.x * 256 + threadIdx.x;
    if (i < n4) {
        float4 f = ((const float4*)in)[i];
        v4h h = {(_Float16)f.x, (_Float16)f.y, (_Float16)f.z, (_Float16)f.w};
        ((v4h*)out)[i] = h;
    }
}

// ---------------- Kernel A: fused QKV-proj + causal attention ---------------
// Phase-1 LDS (halves): sX[64*384]=24576 | sW[2][192*32]=12288  -> 36864 (72KB)
// Phase-2 LDS overlays phase-1: sQ[4096] sK[4096] sVt[4096] sP[4096]
#define SMEM_TOTAL 36864
#define SW_OFF     24576
#define SW_BUFH    6144      // 192*32 halves per buffer

__global__ __launch_bounds__(128) void attn_fused_kernel(
    const _Float16* __restrict__ xh,     // [B, 64, 384] f16
    const _Float16* __restrict__ wqkvh,  // [1152, 384]  f16
    _Float16* __restrict__ attn_out)     // [B, 64, 384] f16
{
    __shared__ _Float16 smem[SMEM_TOTAL];
    _Float16* sX  = smem;                 // phase 1: x slab [64][384]
    _Float16* sWb = smem + SW_OFF;        // phase 1: W tiles, 2 buffers
    _Float16* sQ  = smem;                 // phase 2 overlays
    _Float16* sK  = smem + 4096;
    _Float16* sVt = smem + 8192;
    _Float16* sP  = smem + 12288;

    const int b    = blockIdx.x / HN;
    const int h    = blockIdx.x % HN;
    const int tid  = threadIdx.x;
    const int wave = tid >> 5;
    const int lane = tid & 31;

    const _Float16* xb = xh + (size_t)b * TSEQ * N_EMBD;
    const _Float16* wh = wqkvh + (size_t)h * HS * N_EMBD;

    v8f qacc[4], kacc[4], vacc[4];
    const v8f z8 = {0.f, 0.f, 0.f, 0.f, 0.f, 0.f, 0.f, 0.f};
#pragma unroll
    for (int j = 0; j < 4; ++j) { qacc[j] = z8; kacc[j] = z8; vacc[j] = z8; }

    const unsigned sx_lds = lds_addr_of(sX);
    const unsigned sw_lds = lds_addr_of(sWb);

    // Prologue: whole x slab (64x384) + first weight chunk, all async.
    if (wave == 0) {
        tdm_load_tile_f16(xb, sx_lds, N_EMBD, TSEQ, N_EMBD, TSEQ, N_EMBD);
#pragma unroll
        for (int g = 0; g < 3; ++g)
            tdm_load_tile_f16(wh + (size_t)g * N_EMBD * N_EMBD, /* kc=0 */
                              sw_lds + (unsigned)(g * 64 * 32 * 2),
                              32, 64, N_EMBD, TSEQ, N_EMBD);
    }

    // ---- Phase 1: Q/K/V = x[b] (64x384) * Wqkv_head^T, pipelined chunks ----
    for (int ic = 0; ic < NCHUNK; ++ic) {
        const int kc = ic * 32;
        if (wave == 0) {
            if (ic + 1 < NCHUNK) {
                // Stream next weight chunk into the other buffer, then wait
                // for everything except those 3 in-flight tiles.
                const unsigned dst = sw_lds + (unsigned)(((ic + 1) & 1)
                                     * SW_BUFH * 2);
#pragma unroll
                for (int g = 0; g < 3; ++g)
                    tdm_load_tile_f16(wh + (size_t)g * N_EMBD * N_EMBD
                                         + (kc + 32),
                                      dst + (unsigned)(g * 64 * 32 * 2),
                                      32, 64, N_EMBD, TSEQ, N_EMBD);
                __builtin_amdgcn_s_wait_tensorcnt(3);
            } else {
                __builtin_amdgcn_s_wait_tensorcnt(0);
            }
        }
        __syncthreads();

        const _Float16* sW = sWb + (ic & 1) * SW_BUFH;
        v16h a = load_frag_a(sX, 16 * wave, kc, N_EMBD);
#pragma unroll
        for (int j = 0; j < 4; ++j) {
            qacc[j] = wmma_f16(a, load_frag_b(sW, 16 * j, 0, 32), qacc[j]);
            kacc[j] = wmma_f16(a, load_frag_b(sW, 64 + 16 * j, 0, 32), kacc[j]);
            vacc[j] = wmma_f16(a, load_frag_b(sW, 128 + 16 * j, 0, 32), vacc[j]);
        }
        __syncthreads();
    }

    // Spill accumulators (C layout) to LDS as f16 (overlays phase-1 buffers);
    // V transposed so PV's B operand is n-major.
    {
        const int col  = lane & 15;
        const int row0 = 16 * wave + ((lane >> 4) << 3);
#pragma unroll
        for (int j = 0; j < 4; ++j) {
#pragma unroll
            for (int v = 0; v < 8; ++v) {
                int m = row0 + v;
                int d = 16 * j + col;
                sQ[m * HS + d]    = (_Float16)qacc[j][v];
                sK[m * HS + d]    = (_Float16)kacc[j][v];
                sVt[d * TSEQ + m] = (_Float16)vacc[j][v];
            }
        }
    }
    __syncthreads();

    // ---- Phase 2: S = scale*Q K^T (causal), P = softmax(S), O = P V ----
    const float scale = 0.125f;  // 1/sqrt(64)
    v8f s[4];
#pragma unroll
    for (int j = 0; j < 4; ++j) s[j] = z8;

    v16h q0 = load_frag_a(sQ, 16 * wave, 0, HS);
    v16h q1 = load_frag_a(sQ, 16 * wave, 32, HS);
#pragma unroll
    for (int j = 0; j < 4; ++j) {
        s[j] = wmma_f16(q0, load_frag_b(sK, 16 * j, 0, HS), s[j]);
        s[j] = wmma_f16(q1, load_frag_b(sK, 16 * j, 32, HS), s[j]);
    }

    // Row m of the C tile: VGPR (m%8), lanes [16*(m/8) .. +15].
    {
        const int coln = lane & 15;
        const int half = (lane >> 4) << 3;  // 0 or 8
#pragma unroll
        for (int v = 0; v < 8; ++v) {
            int m = 16 * wave + half + v;
            float mx = -1e30f;
#pragma unroll
            for (int j = 0; j < 4; ++j) {
                int n = 16 * j + coln;
                float val = s[j][v] * scale;
                if (n > m) val = -1e30f;
                s[j][v] = val;
                mx = fmaxf(mx, val);
            }
#pragma unroll
            for (int off = 1; off < 16; off <<= 1)
                mx = fmaxf(mx, __shfl_xor(mx, off, 32));
            float sum = 0.f;
#pragma unroll
            for (int j = 0; j < 4; ++j) {
                float e = __expf(s[j][v] - mx);
                s[j][v] = e;
                sum += e;
            }
#pragma unroll
            for (int off = 1; off < 16; off <<= 1)
                sum += __shfl_xor(sum, off, 32);
            float inv = 1.0f / sum;
#pragma unroll
            for (int j = 0; j < 4; ++j) {
                int n = 16 * j + coln;
                sP[(16 * wave + half + v) * TSEQ + n] = (_Float16)(s[j][v] * inv);
            }
        }
    }
    // sP rows are wave-private; same-wave DS store->load is in-order.

    v8f o[4];
#pragma unroll
    for (int j = 0; j < 4; ++j) o[j] = z8;
    v16h p0 = load_frag_a(sP, 16 * wave, 0, TSEQ);
    v16h p1 = load_frag_a(sP, 16 * wave, 32, TSEQ);
#pragma unroll
    for (int j = 0; j < 4; ++j) {
        o[j] = wmma_f16(p0, load_frag_b(sVt, 16 * j, 0, TSEQ), o[j]);
        o[j] = wmma_f16(p1, load_frag_b(sVt, 16 * j, 32, TSEQ), o[j]);
    }

    // Store O rows into attn_out[b, t, h*64 + d] (f16).
    _Float16* outb = attn_out + (size_t)b * TSEQ * N_EMBD + h * HS;
    {
        const int col  = lane & 15;
        const int row0 = 16 * wave + ((lane >> 4) << 3);
#pragma unroll
        for (int j = 0; j < 4; ++j) {
#pragma unroll
            for (int v = 0; v < 8; ++v)
                outb[(size_t)(row0 + v) * N_EMBD + 16 * j + col] =
                    (_Float16)o[j][v];
        }
    }
}

// ---------------- Kernel B: C = A * Bw^T + bias (WMMA, pipelined TDM) -------
// A: [M,K] f16 row-major; Bw: [N,K] f16 row-major; C: [M,N] f32.
__global__ __launch_bounds__(256) void gemm_nt_kernel(
    const _Float16* __restrict__ A, const _Float16* __restrict__ Bw,
    const float* __restrict__ bias, float* __restrict__ C,
    int M, int N, int K)
{
    __shared__ _Float16 sA[2][128 * 32];
    __shared__ _Float16 sB[2][64 * 32];

    const int tile_m = blockIdx.x * 128;
    const int tile_n = blockIdx.y * 64;
    const int tid  = threadIdx.x;
    const int wave = tid >> 5;
    const int lane = tid & 31;

    v8f acc[4];
    const v8f z8 = {0.f, 0.f, 0.f, 0.f, 0.f, 0.f, 0.f, 0.f};
#pragma unroll
    for (int j = 0; j < 4; ++j) acc[j] = z8;

    const _Float16* Arow = A + (size_t)tile_m * K;
    const _Float16* Brow = Bw + (size_t)tile_n * K;
    const int nchunk = K / 32;

    if (wave == 0) {
        tdm_load_tile_f16(Arow, lds_addr_of(sA[0]), 32, 128,
                          (unsigned)K, (unsigned)M, (unsigned)K);
        tdm_load_tile_f16(Brow, lds_addr_of(sB[0]), 32, 64,
                          (unsigned)K, (unsigned)N, (unsigned)K);
    }

    for (int ic = 0; ic < nchunk; ++ic) {
        if (wave == 0) {
            if (ic + 1 < nchunk) {
                const int kn = (ic + 1) * 32;
                const int nb = (ic + 1) & 1;
                tdm_load_tile_f16(Arow + kn, lds_addr_of(sA[nb]), 32, 128,
                                  (unsigned)K, (unsigned)M, (unsigned)K);
                tdm_load_tile_f16(Brow + kn, lds_addr_of(sB[nb]), 32, 64,
                                  (unsigned)K, (unsigned)N, (unsigned)K);
                __builtin_amdgcn_s_wait_tensorcnt(2);
            } else {
                __builtin_amdgcn_s_wait_tensorcnt(0);
            }
        }
        __syncthreads();

        const int cb = ic & 1;
        v16h a = load_frag_a(sA[cb], 16 * wave, 0, 32);
#pragma unroll
        for (int j = 0; j < 4; ++j)
            acc[j] = wmma_f16(a, load_frag_b(sB[cb], 16 * j, 0, 32), acc[j]);
        __syncthreads();
    }

    const int colq = lane & 15;
    const int row0 = tile_m + 16 * wave + ((lane >> 4) << 3);
#pragma unroll
    for (int j = 0; j < 4; ++j) {
        int col = tile_n + 16 * j + colq;
        float bv = bias ? bias[col] : 0.f;
#pragma unroll
        for (int v = 0; v < 8; ++v)
            C[(size_t)(row0 + v) * N + col] = acc[j][v] + bv;
    }
}

extern "C" void kernel_launch(void* const* d_in, const int* in_sizes, int n_in,
                              void* d_out, int out_size, void* d_ws, size_t ws_size,
                              hipStream_t stream) {
    const float* x    = (const float*)d_in[0];  // [2048, 64, 384]
    const float* Wqkv = (const float*)d_in[1];  // [1152, 384]
    const float* Wp   = (const float*)d_in[2];  // [384, 384]
    const float* bp   = (const float*)d_in[3];  // [384]
    float* out = (float*)d_out;                 // [2048, 64, 384] f32

    const int B = 2048;
    const int M = B * TSEQ;                     // 131072
    const int nX  = M * N_EMBD;                 // 50,331,648
    const int nWq = 3 * N_EMBD * N_EMBD;        // 442,368
    const int nWp = N_EMBD * N_EMBD;            // 147,456

    // Workspace layout (f16).
    char* ws = (char*)d_ws;
    _Float16* xh    = (_Float16*)ws;                         // ~100.7 MB
    _Float16* wqkvh = (_Float16*)(ws + ((size_t)nX * 2));
    _Float16* wph   = (_Float16*)(ws + ((size_t)nX * 2) + (size_t)nWq * 2);
    _Float16* attnh = (_Float16*)(ws + ((size_t)nX * 2) + (size_t)nWq * 2
                                     + (size_t)nWp * 2 + 256);

    // Pass 0: f32 -> f16 (bandwidth-bound one-shot; all sizes divisible by 4).
    cvt_f16_kernel<<<(nX / 4 + 255) / 256, 256, 0, stream>>>(x, xh, nX / 4);
    cvt_f16_kernel<<<(nWq / 4 + 255) / 256, 256, 0, stream>>>(Wqkv, wqkvh, nWq / 4);
    cvt_f16_kernel<<<(nWp / 4 + 255) / 256, 256, 0, stream>>>(Wp, wph, nWp / 4);

    // Kernel A: fused QKV projection + attention, one block per (b, h).
    attn_fused_kernel<<<B * HN, 128, 0, stream>>>(xh, wqkvh, attnh);

    // Kernel B: output projection [M,384] x [384,384]^T + bias.
    dim3 grid(M / 128, N_EMBD / 64);
    gemm_nt_kernel<<<grid, 256, 0, stream>>>(attnh, wph, bp, out, M, N_EMBD, N_EMBD);
}
